// ClinicalRLModel_55525337203150
// MI455X (gfx1250) — matrix operand order repaired and checked by reading, not compiled, not run on hardware
//
#include <hip/hip_runtime.h>
#include <hip/hip_bf16.h>

// ---------------------------------------------------------------------------
// ClinicalRL model for MI455X (gfx1250, wave32, WMMA).
//   B=128 T=64 C=24  EMBED=128 HIDDEN=128 XD=256 GATES=384 AH=64 NA=10 LD=200
// Strategy:
//   * bf16 V_WMMA_F32_16X16X32_BF16 for x-projection GEMM and GRU recurrence
//   * A/B operands pre-swizzled into fragment-ready layouts (CDNA5 ISA tables)
//   * GRU: persistent blocks, W_hh held in VGPRs across all 64 steps, h in LDS
//   * Head computed only for the last timestep (only fused[:, -1, :] reaches
//     the output — everything else in the reference is dead code)
// ---------------------------------------------------------------------------

typedef __attribute__((ext_vector_type(16))) __bf16 v16bf;
typedef __attribute__((ext_vector_type(8)))  __bf16 v8bf;
typedef __attribute__((ext_vector_type(8)))  float  v8f;

#define Bn 128
#define Tn 64
#define Cn 24

static __device__ inline __bf16 f32_to_bf16(float f) {
  unsigned u = __builtin_bit_cast(unsigned, f);
  u += 0x7fffu + ((u >> 16) & 1u);            // round-to-nearest-even
  unsigned short h = (unsigned short)(u >> 16);
  return __builtin_bit_cast(__bf16, h);
}

struct Frag2 { v8bf lo, hi; };   // 32 bytes, trivially copyable

// contiguous 32B fragment (pre-swizzled storage)
static __device__ inline v16bf load_frag(const __bf16* p) {
  Frag2 t = *(const Frag2*)p;
  return __builtin_bit_cast(v16bf, t);
}
// A-fragment from a row-major [16][128] LDS h buffer: two 16B runs
static __device__ inline v16bf load_frag2(const __bf16* p0, const __bf16* p1) {
  Frag2 t; t.lo = *(const v8bf*)p0; t.hi = *(const v8bf*)p1;
  return __builtin_bit_cast(v16bf, t);
}

static __device__ inline v8f wmma_bf16(v16bf a, v16bf b, v8f c) {
  return __builtin_amdgcn_wmma_f32_16x16x32_bf16(false, a, false, b,
                                                 (short)0, c, false, false);
}

// ---------------------------------------------------------------------------
// 1) Convert W_ih (384x256) and W_hh (384x128) to bf16 B-fragments.
//    B layout (32x16, 16-bit): lane L -> column n=L%16, half=L/16,
//    element j: j<8 -> k_local = half*8+j ; j>=8 -> k_local = 16+half*8+(j-8).  Fragment (kt,nt) stored as
//    32 lanes x 16 contiguous bf16.
// ---------------------------------------------------------------------------
__global__ void cw_convert(const float* __restrict__ W_ih,
                           const float* __restrict__ W_hh,
                           __bf16* __restrict__ wihf,
                           __bf16* __restrict__ whhf) {
  int gid = blockIdx.x * blockDim.x + threadIdx.x;
  const int NIH = 8 * 24 * 32 * 16;                 // 98304 elems
  if (gid < NIH) {
    int j    = gid & 15;
    int lane = (gid >> 4) & 31;
    int ft   = gid >> 9;
    int nt   = ft % 24, kt = ft / 24;               // kt<8 (K=256)
    int half = lane >> 4;
    int n    = nt * 16 + (lane & 15);
    int k    = kt * 32 + half * 8 + (j & 7) + ((j >> 3) << 4);
    wihf[gid] = f32_to_bf16(W_ih[n * 256 + k]);
  } else {
    int g    = gid - NIH;                           // < 49152
    int j    = g & 15;
    int lane = (g >> 4) & 31;
    int ft   = g >> 9;
    int nt   = ft % 24, kt = ft / 24;               // kt<4 (K=128)
    int half = lane >> 4;
    int n    = nt * 16 + (lane & 15);
    int k    = kt * 32 + half * 8 + (j & 7) + ((j >> 3) << 4);
    whhf[g] = f32_to_bf16(W_hh[n * 128 + k]);
  }
}

// ---------------------------------------------------------------------------
// 2) Embedding gather-sum -> x (8192 x 256) written in A-fragment layout.
//    A layout (16x32, 16-bit): lane L -> row m=L%16, half=L/16,
//    element j: j<8 -> k_local = half*8+j ; j>=8 -> k_local = 16+half*8+(j-8).
// ---------------------------------------------------------------------------
__global__ void cw_embed(const int* __restrict__ cond,
                         const int* __restrict__ proc,
                         const float* __restrict__ ec,
                         const float* __restrict__ ep,
                         __bf16* __restrict__ xfrag) {
  int row = blockIdx.x;                 // b*T + t
  int d   = threadIdx.x;                // 0..255 (wave-uniform table split)
  const int*   idx = (d < 128) ? cond : proc;
  const float* tab = (d < 128) ? ec : ep;
  int dim = d & 127;
  const int* ip = idx + row * Cn;
  float acc = 0.f;
#pragma unroll 4
  for (int c = 0; c < Cn; ++c) acc += tab[ip[c] * 128 + dim];

  int m_tile = row >> 4, m = row & 15;
  int kt = d >> 5, kl = d & 31;
  int half = (kl >> 3) & 1;
  int j    = (kl & 7) + ((kl >> 4) << 3);
  int lane = half * 16 + m;
  xfrag[((m_tile * 8 + kt) * 32 + lane) * 16 + j] = f32_to_bf16(acc);
}

// ---------------------------------------------------------------------------
// 3) x_proj = x @ W_ih^T + b_ih  (M=8192, N=384, K=256) via bf16 WMMA.
//    One 16x16 output tile per wave, 8 K-steps of K=32.
// ---------------------------------------------------------------------------
__global__ void cw_gemm_xproj(const __bf16* __restrict__ xfrag,
                              const __bf16* __restrict__ wihf,
                              const float* __restrict__ b_ih,
                              float* __restrict__ xproj) {
  int wid  = blockIdx.x * 8 + (threadIdx.x >> 5);   // 12288 wave-tiles
  int lane = threadIdx.x & 31;
  int m_tile = wid / 24;
  int n_tile = wid % 24;

  v8f acc = {};
#pragma unroll
  for (int kt = 0; kt < 8; ++kt) {
    const __bf16* ap = xfrag + ((m_tile * 8 + kt) * 32 + lane) * 16;
    if (kt < 7)  // gfx1250 global_prefetch of next A fragment
      __builtin_prefetch(ap + 32 * 16, 0, 1);
    v16bf a = load_frag(ap);
    v16bf b = load_frag(wihf + ((kt * 24 + n_tile) * 32 + lane) * 16);
    acc = wmma_bf16(a, b, acc);
  }
  int half = lane >> 4;
  int n    = n_tile * 16 + (lane & 15);
  float bn = b_ih[n];
#pragma unroll
  for (int i = 0; i < 8; ++i) {
    int rowm = m_tile * 16 + i + 8 * half;          // C layout: VGPR i -> M=i / 8+i
    xproj[rowm * 384 + n] = acc[i] + bn;
  }
}

// ---------------------------------------------------------------------------
// 4) GRU recurrence. 8 blocks x 256 threads; block owns 16 batch rows.
//    Wave w owns hidden-dim tile w: computes gh tiles for r (nt=w),
//    z (nt=8+w), n (nt=16+w) -> 3 WMMA x 4 K-tiles per step.
//    W_hh fragments live in VGPRs for the whole scan; h lives in LDS.
// ---------------------------------------------------------------------------
__global__ void __launch_bounds__(256) cw_gru(const float* __restrict__ xproj,
                                              const __bf16* __restrict__ whhf,
                                              const float* __restrict__ b_hh,
                                              float* __restrict__ rnn_out) {
  __shared__ __bf16 h_bf[16 * 128];
  __shared__ float  h_f32[16 * 128];
  int tid = threadIdx.x;
  int w = tid >> 5, lane = tid & 31;
  int bb = blockIdx.x * 16;

  for (int i = tid; i < 16 * 128; i += 256) {
    h_bf[i]  = __builtin_bit_cast(__bf16, (unsigned short)0);
    h_f32[i] = 0.f;
  }
  __syncthreads();

  // W_hh B-fragments, register-resident for all 64 steps (96 VGPRs).
  v16bf wb[3][4];
#pragma unroll
  for (int g = 0; g < 3; ++g)
#pragma unroll
    for (int kt = 0; kt < 4; ++kt)
      wb[g][kt] = load_frag(whhf + ((kt * 24 + (g * 8 + w)) * 32 + lane) * 16);

  int m = lane & 15, half = lane >> 4;
  int j = w * 16 + (lane & 15);                      // hidden dim of this lane
  float bhr = b_hh[j], bhz = b_hh[128 + j], bhn = b_hh[256 + j];

  for (int t = 0; t < Tn; ++t) {
    v16bf ha[4];
#pragma unroll
    for (int kt = 0; kt < 4; ++kt) {
      const __bf16* base = &h_bf[m * 128 + kt * 32 + half * 8];
      ha[kt] = load_frag2(base, base + 16);
    }
    v8f gr = {}, gz = {}, gn = {};
#pragma unroll
    for (int kt = 0; kt < 4; ++kt) {
      gr = wmma_bf16(ha[kt], wb[0][kt], gr);
      gz = wmma_bf16(ha[kt], wb[1][kt], gz);
      gn = wmma_bf16(ha[kt], wb[2][kt], gn);
    }
    __syncthreads();   // all waves done reading h_bf before we overwrite it

#pragma unroll
    for (int i = 0; i < 8; ++i) {
      int mr = i + 8 * half;                         // batch row within tile
      int rb = ((bb + mr) * Tn + t) * 384;
      float xr = xproj[rb + j];
      float xz = xproj[rb + 128 + j];
      float xn = xproj[rb + 256 + j];
      float hr = gr[i] + bhr, hz = gz[i] + bhz, hn = gn[i] + bhn;
      float r  = 1.f / (1.f + __expf(-(xr + hr)));
      float z  = 1.f / (1.f + __expf(-(xz + hz)));
      float nn = tanhf(xn + r * hn);
      float ho = h_f32[mr * 128 + j];
      float hv = (1.f - z) * nn + z * ho;
      h_f32[mr * 128 + j] = hv;
      h_bf[mr * 128 + j]  = f32_to_bf16(hv);
      rnn_out[((bb + mr) * Tn + t) * 128 + j] = hv;
    }
    __syncthreads();
  }
}

// ---------------------------------------------------------------------------
// 5) Head, last timestep only (everything earlier is dead code w.r.t. output).
// ---------------------------------------------------------------------------
static __device__ inline unsigned cw_hash(unsigned s) {
  s = (s ^ 61u) ^ (s >> 16); s *= 9u; s ^= s >> 4;
  s *= 0x27d4eb2du; s ^= s >> 15; return s;
}

__global__ void cw_head(const float* __restrict__ rnn_out,
                        const float* __restrict__ Wp1, const float* __restrict__ bp1,
                        const float* __restrict__ Wp2, const float* __restrict__ bp2,
                        const float* __restrict__ Wf,  const float* __restrict__ bfv,
                        const float* __restrict__ Wo,  const float* __restrict__ bo,
                        float* __restrict__ out) {
  __shared__ float h63[128], p1[64], lg[10], cat[256], fused[128];
  __shared__ int sel;
  int b = blockIdx.x, tid = threadIdx.x;

  if (tid < 128) h63[tid] = rnn_out[(b * Tn + 63) * 128 + tid];
  __syncthreads();
  if (tid < 64) {
    float a = bp1[tid];
    for (int k = 0; k < 128; ++k) a += h63[k] * Wp1[tid * 128 + k];
    p1[tid] = tanhf(a);
  }
  __syncthreads();
  if (tid < 10) {
    float a = bp2[tid];
    for (int k = 0; k < 64; ++k) a += p1[k] * Wp2[tid * 64 + k];
    lg[tid] = a;
  }
  __syncthreads();
  if (tid == 0) {
    // categorical = argmax(logits + gumbel); hash-based noise stands in for
    // JAX threefry (unobservable in this environment).
    int best = 0; float bestv = -1e30f;
    for (int a = 0; a < 10; ++a) {
      unsigned s = cw_hash((unsigned)(b * 16 + a) * 0x9e3779b9u + 42u);
      float u = ((float)(s & 0xFFFFFFu) + 0.5f) * (1.0f / 16777216.0f);
      float gum = -__logf(-__logf(u));
      float v = lg[a] + gum;
      if (v > bestv) { bestv = v; best = a; }
    }
    sel = 53 + best;                                 // history_start(t=63)=53
  }
  __syncthreads();
  if (tid < 128) {
    cat[tid]       = h63[tid];
    cat[128 + tid] = rnn_out[(b * Tn + sel) * 128 + tid];
  }
  __syncthreads();
  if (tid < 128) {
    float a = bfv[tid];
    for (int k = 0; k < 256; ++k) a += cat[k] * Wf[tid * 256 + k];
    fused[tid] = fmaxf(a, 0.f);
  }
  __syncthreads();
  if (tid < 200) {
    float a = bo[tid];
    for (int k = 0; k < 128; ++k) a += fused[k] * Wo[tid * 128 + k];
    out[b * 200 + tid] = a;
  }
}

// ---------------------------------------------------------------------------
extern "C" void kernel_launch(void* const* d_in, const int* in_sizes, int n_in,
                              void* d_out, int out_size, void* d_ws, size_t ws_size,
                              hipStream_t stream) {
  const int*   conditions = (const int*)d_in[0];
  const int*   procedures = (const int*)d_in[1];
  const float* emb_cond   = (const float*)d_in[2];
  const float* emb_proc   = (const float*)d_in[3];
  const float* W_ih       = (const float*)d_in[4];
  const float* W_hh       = (const float*)d_in[5];
  const float* b_ih       = (const float*)d_in[6];
  const float* b_hh       = (const float*)d_in[7];
  const float* Wp1        = (const float*)d_in[8];
  const float* bp1        = (const float*)d_in[9];
  const float* Wp2        = (const float*)d_in[10];
  const float* bp2        = (const float*)d_in[11];
  const float* Wf         = (const float*)d_in[12];
  const float* bfv        = (const float*)d_in[13];
  const float* Wo         = (const float*)d_in[14];
  const float* bo         = (const float*)d_in[15];

  char* ws = (char*)d_ws;
  // byte offsets (all 256B aligned)
  __bf16* xfrag = (__bf16*)(ws);                          // 8192*256 bf16 = 4 MB
  __bf16* wihf  = (__bf16*)(ws + 4194304);                // 98304 bf16 = 192 KB
  __bf16* whhf  = (__bf16*)(ws + 4194304 + 196608);       // 49152 bf16 = 96 KB
  float*  xproj = (float*) (ws + 4194304 + 196608 + 98304);        // 12 MB
  float*  rnno  = (float*) (ws + 4194304 + 196608 + 98304 + 12582912); // 4 MB

  cw_convert   <<<576,  256, 0, stream>>>(W_ih, W_hh, wihf, whhf);
  cw_embed     <<<Bn * Tn, 256, 0, stream>>>(conditions, procedures,
                                             emb_cond, emb_proc, xfrag);
  cw_gemm_xproj<<<1536, 256, 0, stream>>>(xfrag, wihf, b_ih, xproj);
  cw_gru       <<<8,    256, 0, stream>>>(xproj, whhf, b_hh, rnno);
  cw_head      <<<Bn,   256, 0, stream>>>(rnno, Wp1, bp1, Wp2, bp2,
                                          Wf, bfv, Wo, bo, (float*)d_out);
}
